// xFormersWithBuffer_41171556499847
// MI455X (gfx1250) — matrix operand order repaired and checked by reading, not compiled, not run on hardware
//
#include <hip/hip_runtime.h>
#include <hip/hip_bf16.h>

// ---------------------------------------------------------------------------
// Paged-KV block-diagonal causal attention for MI455X (gfx1250, wave32).
// bf16 WMMA (v_wmma_f32_16x16x32_bf16) with f32 accumulate, flash-attention
// online softmax (exp2 domain), software-pipelined gathered K/V loads.
// ---------------------------------------------------------------------------

typedef __attribute__((ext_vector_type(16))) __bf16 v16bf;
typedef __attribute__((ext_vector_type(8)))  float  v8f;

#define NUM_SEQS 4
#define Q_LEN    512
#define KV_LEN   4096
#define NH       32
#define DH       128
#define CACHE_N  32768
#define SCALE_F  0.08838834764831845f
#define LOG2E_F  1.4426950408889634f
#define C2_F     (SCALE_F * LOG2E_F)   // log2-domain score scale
#define QOFF     (KV_LEN - Q_LEN)      // 3584

#define KT    64      // keys per tile
#define KSTR  136     // K tile LDS row stride (bf16 elems), 272B rows (16B aligned)
#define VSTR  72      // V^T LDS row stride, 144B rows (16B aligned)
#define PSTR  72      // P LDS row stride

union Frag16 {
    v16bf  v;
    uint4  u[2];
    __bf16 e[16];
};

static __device__ __forceinline__ float fast_exp2(float x) {
#if __has_builtin(__builtin_amdgcn_exp2f)
    return __builtin_amdgcn_exp2f(x);      // raw v_exp_f32
#else
    return exp2f(x);
#endif
}

// ---------------------------------------------------------------------------
// inverse-map prep kernels: inv[slot] = new-token index j if alloc_idx[j]==slot
// ---------------------------------------------------------------------------
__global__ void init_inv_kernel(int* __restrict__ inv, int n) {
    int i = blockIdx.x * 256 + threadIdx.x;
    if (i < n) inv[i] = -1;
}

__global__ void fill_inv_kernel(const int* __restrict__ alloc_idx,
                                int* __restrict__ inv, int n) {
    int i = blockIdx.x * 256 + threadIdx.x;
    if (i < n) inv[alloc_idx[i]] = i;
}

// ---------------------------------------------------------------------------
// Flash attention kernel. Grid: 512 blocks = 4 seq * 32 heads * 4 q-blocks.
// Block: 256 threads = 8 waves; wave w owns q rows [qb*128 + w*16, +16).
// ---------------------------------------------------------------------------
__global__ __launch_bounds__(256)
void paged_attn_kernel(const float* __restrict__ q,
                       const float* __restrict__ knew,
                       const float* __restrict__ vnew,
                       const float* __restrict__ k_cache,
                       const float* __restrict__ v_cache,
                       const int*   __restrict__ ctx_idx,
                       const int*   __restrict__ inv,
                       float*       __restrict__ out)
{
    __shared__ __bf16 Klds[KT][KSTR];        // K tile, row-major [token][d]
    __shared__ __bf16 Vt[DH][VSTR];          // V tile, transposed [d][token]
    __shared__ __bf16 Plds[8][16][PSTR];     // per-wave P tile [row][key]

    const int tid    = threadIdx.x;
    const int wave   = tid >> 5;
    const int lane   = tid & 31;
    const int lanelo = lane & 15;
    const int hb     = lane >> 4;            // lane-half select (K split)

    const int b  = blockIdx.x;
    const int qb = b & 3;                    // q-block within sequence
    const int h  = (b >> 2) & 31;            // head
    const int s  = b >> 7;                   // sequence

    // ---- Q A-fragments (16x32 per kc), held in registers for whole loop ----
    // A layout: lane holds row M=lane%16; elems 0..7 = K(d) = kc*32+hb*8+i,
    //           elems 8..15 = kc*32+16+hb*8+i.
    const int qrow_local = qb * 128 + wave * 16 + lanelo;   // 0..511
    const float* qrow = q + (size_t)(s * Q_LEN + qrow_local) * (NH * DH) + h * DH;
    Frag16 qf[4];
#pragma unroll
    for (int kc = 0; kc < 4; ++kc) {
        const int d0 = kc * 32 + hb * 8;
#pragma unroll
        for (int i = 0; i < 8; ++i) {
            qf[kc].e[i]     = (__bf16)qrow[d0 + i];
            qf[kc].e[8 + i] = (__bf16)qrow[d0 + 16 + i];
        }
    }

    // C/D layout: VGPR r holds row M = r + hb*8, col N = lanelo (per N-tile).
    const int qbase = QOFF + qb * 128 + wave * 16 + hb * 8;  // qpos of row r=0

    v8f acc[8];                              // 16x128 output tile (8 N-tiles)
    float mrun[8], lrun[8];                  // mrun kept in log2 domain
#pragma unroll
    for (int nt = 0; nt < 8; ++nt) acc[nt] = (v8f){};
#pragma unroll
    for (int r = 0; r < 8; ++r) { mrun[r] = -1e30f; lrun[r] = 0.0f; }

    const int nkt      = 58 + 2 * qb;        // tiles needed before causal cutoff
    const int ctx_base = s * KV_LEN;

    // ---- gathered global load of one K/V tile into bf16 staging registers --
    __bf16 kst[8][4], vst[8][4];
    auto load_tile = [&](int kt) {
#pragma unroll
        for (int it = 0; it < 8; ++it) {
            const int c   = tid + it * 256;
            const int tok = c >> 5;          // 0..63
            const int ch  = c & 31;          // float4 chunk of D
            const int g    = ctx_base + kt * KT + tok;
            const int slot = ctx_idx[g];
            const int j    = inv[slot];
            const float* kb = (j >= 0) ? (knew    + (size_t)j    * (NH * DH))
                                       : (k_cache + (size_t)slot * (NH * DH));
            const float* vb = (j >= 0) ? (vnew    + (size_t)j    * (NH * DH))
                                       : (v_cache + (size_t)slot * (NH * DH));
            const float4 kd = *(const float4*)(kb + h * DH + ch * 4);
            const float4 vd = *(const float4*)(vb + h * DH + ch * 4);
            kst[it][0] = (__bf16)kd.x; kst[it][1] = (__bf16)kd.y;
            kst[it][2] = (__bf16)kd.z; kst[it][3] = (__bf16)kd.w;
            vst[it][0] = (__bf16)vd.x; vst[it][1] = (__bf16)vd.y;
            vst[it][2] = (__bf16)vd.z; vst[it][3] = (__bf16)vd.w;
        }
    };

    auto store_tile = [&]() {
#pragma unroll
        for (int it = 0; it < 8; ++it) {
            const int c   = tid + it * 256;
            const int tok = c >> 5;
            const int ch  = c & 31;
            __bf16* kp = &Klds[tok][ch * 4];
            kp[0] = kst[it][0]; kp[1] = kst[it][1];
            kp[2] = kst[it][2]; kp[3] = kst[it][3];
            const int d0 = ch * 4;
            Vt[d0 + 0][tok] = vst[it][0];
            Vt[d0 + 1][tok] = vst[it][1];
            Vt[d0 + 2][tok] = vst[it][2];
            Vt[d0 + 3][tok] = vst[it][3];
        }
    };

    load_tile(0);                            // prologue

    for (int kt = 0; kt < nkt; ++kt) {
        store_tile();                        // staging regs -> LDS
        __syncthreads();                     // tile visible to all waves

        if (kt + 1 < nkt)                    // prefetch next tile's gathers;
            load_tile(kt + 1);               // VMEM latency hides under WMMAs

        // ---- S = Q * K^T  (16 x 64), 4 N-tiles x 4 K-chunks of WMMA --------
        v8f sc[4];
#pragma unroll
        for (int nt = 0; nt < 4; ++nt) {
            v8f c = (v8f){};
#pragma unroll
            for (int kc = 0; kc < 4; ++kc) {
                // B layout: lane holds col N=lanelo of key nt*16+lanelo,
                // K(d) = kc*32 + hb*8 + 0..7, then kc*32 + 16 + hb*8 + 0..7.
                Frag16 bfr;
                const __bf16* bp = &Klds[nt * 16 + lanelo][kc * 32 + hb * 8];
                bfr.u[0] = *(const uint4*)(bp);
                bfr.u[1] = *(const uint4*)(bp + 16);
                c = __builtin_amdgcn_wmma_f32_16x16x32_bf16(
                        false, qf[kc].v, false, bfr.v, (short)0, c, false, false);
            }
            sc[nt] = c;
        }

        // ---- log2-domain scale + causal mask + row max ---------------------
        float mloc[8];
#pragma unroll
        for (int r = 0; r < 8; ++r) mloc[r] = -1e30f;
        const int ktb = kt * KT;
#pragma unroll
        for (int nt = 0; nt < 4; ++nt) {
            const int kpos = ktb + nt * 16 + lanelo;
#pragma unroll
            for (int r = 0; r < 8; ++r) {
                float x = sc[nt][r] * C2_F;          // SCALE * log2(e)
                if (kpos > qbase + r) x = -1e30f;
                sc[nt][r] = x;
                mloc[r] = fmaxf(mloc[r], x);
            }
        }
#pragma unroll
        for (int r = 0; r < 8; ++r) {
#pragma unroll
            for (int off = 1; off < 16; off <<= 1)
                mloc[r] = fmaxf(mloc[r], __shfl_xor(mloc[r], off, 16));
        }

        float alpha[8], lloc[8];
#pragma unroll
        for (int r = 0; r < 8; ++r) {
            const float mn = fmaxf(mrun[r], mloc[r]);
            alpha[r] = fast_exp2(mrun[r] - mn);      // exp2: raw v_exp_f32
            mrun[r]  = mn;
            lloc[r]  = 0.0f;
        }

        // ---- P = exp2(S - m); stage to LDS row-major for A-layout reload ---
#pragma unroll
        for (int nt = 0; nt < 4; ++nt) {
#pragma unroll
            for (int r = 0; r < 8; ++r) {
                const float p = fast_exp2(sc[nt][r] - mrun[r]);
                lloc[r] += p;
                Plds[wave][r + hb * 8][nt * 16 + lanelo] = (__bf16)p;
            }
        }
#pragma unroll
        for (int r = 0; r < 8; ++r) {
#pragma unroll
            for (int off = 1; off < 16; off <<= 1)
                lloc[r] += __shfl_xor(lloc[r], off, 16);
            lrun[r] = lrun[r] * alpha[r] + lloc[r];
        }

        // ---- rescale accumulators ------------------------------------------
#pragma unroll
        for (int nt = 0; nt < 8; ++nt)
#pragma unroll
            for (int r = 0; r < 8; ++r)
                acc[nt][r] *= alpha[r];

        // ---- O += P * V  (16 x 128), 2 K-chunks x 8 N-tiles of WMMA --------
#pragma unroll
        for (int kc = 0; kc < 2; ++kc) {
            // A layout from Plds: row M=lanelo; elems 0..7 = key kc*32+hb*8+i,
            // elems 8..15 = kc*32+16+hb*8+i.
            Frag16 pa;
            const __bf16* pp = &Plds[wave][lanelo][kc * 32 + hb * 8];
            pa.u[0] = *(const uint4*)(pp);
            pa.u[1] = *(const uint4*)(pp + 16);
#pragma unroll
            for (int nt = 0; nt < 8; ++nt) {
                // B layout from Vt: lane holds col N(d)=nt*16+lanelo,
                // K(keys) = kc*32 + hb*8 + 0..7, then kc*32 + 16 + hb*8 + 0..7.
                Frag16 vbf;
                const __bf16* vp = &Vt[nt * 16 + lanelo][kc * 32 + hb * 8];
                vbf.u[0] = *(const uint4*)(vp);
                vbf.u[1] = *(const uint4*)(vp + 16);
                acc[nt] = __builtin_amdgcn_wmma_f32_16x16x32_bf16(
                        false, pa.v, false, vbf.v, (short)0, acc[nt], false, false);
            }
        }

        __syncthreads();                     // readers done before next store
    }

    // ---- epilogue: O /= l, write f32 output [token][h*128 + d] -------------
#pragma unroll
    for (int r = 0; r < 8; ++r) {
        const float invl = 1.0f / lrun[r];
        const int tok = s * Q_LEN + qb * 128 + wave * 16 + r + hb * 8;
        float* orow = out + (size_t)tok * (NH * DH) + h * DH;
#pragma unroll
        for (int nt = 0; nt < 8; ++nt)
            orow[nt * 16 + lanelo] = acc[nt][r] * invl;
    }
}

// ---------------------------------------------------------------------------
extern "C" void kernel_launch(void* const* d_in, const int* in_sizes, int n_in,
                              void* d_out, int out_size, void* d_ws, size_t ws_size,
                              hipStream_t stream) {
    const float* q       = (const float*)d_in[0];
    const float* k       = (const float*)d_in[1];
    const float* v       = (const float*)d_in[2];
    const float* k_cache = (const float*)d_in[3];
    const float* v_cache = (const float*)d_in[4];
    const int*   alloc   = (const int*)d_in[5];
    const int*   ctx     = (const int*)d_in[6];
    float*       out     = (float*)d_out;

    int* inv = (int*)d_ws;                   // 32768 * 4B = 128 KB scratch

    init_inv_kernel<<<CACHE_N / 256, 256, 0, stream>>>(inv, CACHE_N);
    fill_inv_kernel<<<(NUM_SEQS * Q_LEN) / 256, 256, 0, stream>>>(
        alloc, inv, NUM_SEQS * Q_LEN);

    // grid: seq(4) * head(32) * qblock(4) = 512 workgroups of 256 threads
    paged_attn_kernel<<<NUM_SEQS * NH * 4, 256, 0, stream>>>(
        q, k, v, k_cache, v_cache, ctx, inv, out);
}
